// QWenBlock_46162308498122
// MI455X (gfx1250) — compile-verified, matrix operand order checked by reading, not run on hardware
//
#include <hip/hip_runtime.h>

// ---------------------------------------------------------------------------
// Types
// ---------------------------------------------------------------------------
typedef unsigned short u16;
typedef unsigned int   u32;
typedef __bf16 v16bf __attribute__((ext_vector_type(16)));
typedef float  v8f   __attribute__((ext_vector_type(8)));

union Frag {
    v16bf bf;
    uint4 q[2];
};

__device__ __forceinline__ u16 f2bf(float f) {
    u32 u; __builtin_memcpy(&u, &f, 4);
    u += 0x7FFFu + ((u >> 16) & 1u);          // round-to-nearest-even
    return (u16)(u >> 16);
}
__device__ __forceinline__ float bf2f(u16 h) {
    u32 u = ((u32)h) << 16;
    float f; __builtin_memcpy(&f, &u, 4);
    return f;
}

__device__ __forceinline__ v8f wmma_bf16(const Frag& a, const Frag& b, v8f c) {
    // D = A(16x32 bf16) * B(32x16 bf16) + C(16x16 f32)
    return __builtin_amdgcn_wmma_f32_16x16x32_bf16(
        false, a.bf, false, b.bf, (short)0, c, false, false);
}

// --- CDNA5 async global->LDS copy (ASYNCcnt path), via inline asm ----------
// LDS generic pointers carry the LDS byte offset in the low 32 bits
// (aperture scheme).
__device__ __forceinline__ unsigned lds_off(const void* p) {
    return (unsigned)(size_t)p;
}
__device__ __forceinline__ void async_b128(unsigned lds_addr, const u16* gaddr) {
    asm volatile("global_load_async_to_lds_b128 %0, %1, off"
                 :: "v"(lds_addr), "v"((unsigned long long)(size_t)gaddr)
                 : "memory");
}
__device__ __forceinline__ void wait_async0() {
    asm volatile("s_wait_asynccnt 0x0" ::: "memory");
}

#define Tt 2048
#define Hh 4096
#define NHh 32
#define HDd 128
#define FFf 11008

// ---------------------------------------------------------------------------
// Weight convert + transpose: fp32 W[K][N] -> bf16 Wt[N][K]
// ---------------------------------------------------------------------------
__global__ __launch_bounds__(256) void cvt_transpose_kernel(
    const float* __restrict__ W, u16* __restrict__ Wt, int K, int N)
{
    __shared__ float tile[32][33];
    int n0 = blockIdx.x * 32, k0 = blockIdx.y * 32;
    int tx = threadIdx.x, ty = threadIdx.y;   // (32, 8)
    for (int i = 0; i < 4; i++) {
        int kk = ty + i * 8;
        tile[kk][tx] = W[(size_t)(k0 + kk) * N + n0 + tx];
    }
    __syncthreads();
    for (int i = 0; i < 4; i++) {
        int nn = ty + i * 8;
        Wt[(size_t)(n0 + nn) * K + k0 + tx] = f2bf(tile[tx][nn]);
    }
}

// ---------------------------------------------------------------------------
// RMSNorm: fp32 row -> bf16 row
// ---------------------------------------------------------------------------
__global__ __launch_bounds__(256) void rmsnorm_kernel(
    const float* __restrict__ X, const float* __restrict__ S,
    u16* __restrict__ O, int H)
{
    int row = blockIdx.x, tid = threadIdx.x;
    const float* x = X + (size_t)row * H;
    float ss = 0.f;
    for (int i = tid; i < H; i += 256) { float v = x[i]; ss = fmaf(v, v, ss); }
    for (int off = 16; off > 0; off >>= 1) ss += __shfl_xor(ss, off);
    __shared__ float red[8];
    if ((tid & 31) == 0) red[tid >> 5] = ss;
    __syncthreads();
    float tot = 0.f;
    for (int w = 0; w < 8; w++) tot += red[w];
    float inv = rsqrtf(tot / (float)H + 1e-6f);
    u16* o = O + (size_t)row * H;
    for (int i = tid; i < H; i += 256) o[i] = f2bf(x[i] * inv * S[i]);
}

// ---------------------------------------------------------------------------
// Generic bf16 WMMA GEMM: C[M][N] = A[M][K] * Wt[N][K]^T (+bias) (+residual)
// 256 threads = 8 waves; block tile 128x128, K-step 32.
// Double-buffered LDS fed by GLOBAL_LOAD_ASYNC_TO_LDS_B128 (ASYNCcnt).
//   outB != nullptr : store bf16 (optionally +bias)
//   outB == nullptr : store f32 = acc (+bias) (+resid)
// ---------------------------------------------------------------------------
#define G_LD 40   // padded LDS row stride (ushorts) -> conflict-free frag loads

__global__ __launch_bounds__(256) void gemm_bf16_kernel(
    const u16* __restrict__ A, const u16* __restrict__ Bt,
    const float* __restrict__ bias, const float* __restrict__ resid,
    float* __restrict__ outF, u16* __restrict__ outB,
    int M, int N, int K)
{
    __shared__ u16 As[2][128 * G_LD];
    __shared__ u16 Bs[2][128 * G_LD];

    int tid = threadIdx.x;
    int wid = tid >> 5, lane = tid & 31, half = lane >> 4, mr = lane & 15;
    int bm = blockIdx.y * 128, bn = blockIdx.x * 128;
    int wm = (wid & 1) * 64;        // 4 m-tiles per wave
    int wn = (wid >> 1) * 32;       // 2 n-tiles per wave

    v8f acc[4][2] = {};

    // per-thread async-copy lanes: 128 rows x 32 cols tile, 32 B per thread
    int lr = tid >> 1, lc = (tid & 1) * 16;
    const u16* sa0 = A  + (size_t)(bm + lr) * K + lc;
    const u16* sb0 = Bt + (size_t)(bn + lr) * K + lc;
    unsigned aoff[2], boff[2];
    for (int b = 0; b < 2; b++) {
        aoff[b] = lds_off(&As[b][lr * G_LD + lc]);
        boff[b] = lds_off(&Bs[b][lr * G_LD + lc]);
    }

    // prologue: stage 0
    async_b128(aoff[0],      sa0);
    async_b128(aoff[0] + 16, sa0 + 8);
    async_b128(boff[0],      sb0);
    async_b128(boff[0] + 16, sb0 + 8);
    wait_async0();
    __syncthreads();

    int nsteps = K >> 5;
    for (int s = 0; s < nsteps; s++) {
        int cur = s & 1, nxt = cur ^ 1;
        if (s + 1 < nsteps) {           // async-fill next buffer while computing
            const u16* sa = sa0 + (size_t)(s + 1) * 32;
            const u16* sb = sb0 + (size_t)(s + 1) * 32;
            async_b128(aoff[nxt],      sa);
            async_b128(aoff[nxt] + 16, sa + 8);
            async_b128(boff[nxt],      sb);
            async_b128(boff[nxt] + 16, sb + 8);
        }

        Frag af[4], bfr[2];
        for (int i = 0; i < 4; i++) {
            int base = (wm + i * 16 + mr) * G_LD + half * 8;
            af[i].q[0] = *(const uint4*)&As[cur][base];
            af[i].q[1] = *(const uint4*)&As[cur][base + 16];
        }
        for (int j = 0; j < 2; j++) {
            int base = (wn + j * 16 + mr) * G_LD + half * 8;
            bfr[j].q[0] = *(const uint4*)&Bs[cur][base];
            bfr[j].q[1] = *(const uint4*)&Bs[cur][base + 16];
        }
        for (int i = 0; i < 4; i++)
            for (int j = 0; j < 2; j++)
                acc[i][j] = wmma_bf16(af[i], bfr[j], acc[i][j]);

        wait_async0();      // next buffer landed; frag reads of cur are done
        __syncthreads();
    }

    for (int i = 0; i < 4; i++) {
        for (int j = 0; j < 2; j++) {
            int n = bn + wn + j * 16 + mr;
            for (int r = 0; r < 8; r++) {
                int m = bm + wm + i * 16 + r + half * 8;
                float v = acc[i][j][r];
                if (bias) v += bias[n];
                size_t idx = (size_t)m * N + n;
                if (outB) outB[idx] = f2bf(v);
                else      outF[idx] = v + (resid ? resid[idx] : 0.f);
            }
        }
    }
}

// ---------------------------------------------------------------------------
// RoPE, in place on bf16 [T][NH*HD]
// ---------------------------------------------------------------------------
__global__ __launch_bounds__(256) void rope_kernel(
    u16* __restrict__ X, const int* __restrict__ positions)
{
    int idx = blockIdx.x * 256 + threadIdx.x;       // T*NH*(HD/2) threads
    int i = idx & 63;
    int h = (idx >> 6) & (NHh - 1);
    int t = idx >> 11;
    float inv_freq = __expf(-((float)(2 * i) / (float)HDd) * __logf(10000.0f));
    float ang = (float)positions[t] * inv_freq;
    float c, s;
    __sincosf(ang, &s, &c);
    size_t base = (size_t)t * Hh + (size_t)h * HDd + i;
    float x1 = bf2f(X[base]);
    float x2 = bf2f(X[base + 64]);
    X[base]      = f2bf(x1 * c - x2 * s);
    X[base + 64] = f2bf(x2 * c + x1 * s);
}

// ---------------------------------------------------------------------------
// V transpose per head: bf16 [T][NH*HD] -> bf16 [NH][HD][T]
// ---------------------------------------------------------------------------
__global__ __launch_bounds__(256) void transpose_v_kernel(
    const u16* __restrict__ V, u16* __restrict__ Vt)
{
    __shared__ u16 tile[32][33];
    int head = blockIdx.z;
    int t0 = blockIdx.x * 32, d0 = blockIdx.y * 32;
    int tx = threadIdx.x, ty = threadIdx.y;   // (32, 8)
    for (int i = 0; i < 4; i++) {
        int tt = ty + i * 8;
        tile[tt][tx] = V[(size_t)(t0 + tt) * Hh + (size_t)head * HDd + d0 + tx];
    }
    __syncthreads();
    for (int i = 0; i < 4; i++) {
        int dd = ty + i * 8;
        Vt[((size_t)head * HDd + d0 + dd) * Tt + t0 + tx] = tile[tx][dd];
    }
}

// ---------------------------------------------------------------------------
// Flash attention: per (q-block of 128 rows, head).
// 8 waves; each wave owns 16 q-rows. Online softmax; QK^T and P*V via WMMA.
// K/V tiles staged to LDS with async global->LDS copies.
// ---------------------------------------------------------------------------
#define AT_LDK 136
#define AT_LDV 40
#define AT_LDP 40

__global__ __launch_bounds__(256) void attention_kernel(
    const u16* __restrict__ Q, const u16* __restrict__ Kc,
    const u16* __restrict__ Vt, u16* __restrict__ Out)
{
    __shared__ u16 Ks[32 * AT_LDK];       // 32 keys x 128 d
    __shared__ u16 Vs[128 * AT_LDV];      // 128 d x 32 keys (pre-transposed V)
    __shared__ u16 Ps[8 * 16 * AT_LDP];   // per-wave P staging 16x32

    int tid = threadIdx.x;
    int wid = tid >> 5, lane = tid & 31, half = lane >> 4, mr = lane & 15;
    int head = blockIdx.y, qb = blockIdx.x;
    int qbase = qb * 128 + wid * 16;
    const float scaling = 0.08838834764831845f;   // 128^-0.5

    // Q fragments (16 rows x 128 d = 4 fragments), straight from global
    Frag qf[4];
    {
        const u16* qrow = Q + (size_t)(qbase + mr) * Hh + (size_t)head * HDd;
        for (int dc = 0; dc < 4; dc++) {
            qf[dc].q[0] = *(const uint4*)(qrow + dc * 32 + half * 8);
            qf[dc].q[1] = *(const uint4*)(qrow + dc * 32 + 16 + half * 8);
        }
    }

    // async-copy lane mappings
    int kr = tid >> 3, kc = (tid & 7) * 16;    // K tile: 32 rows x 128 d
    unsigned kdst = lds_off(&Ks[kr * AT_LDK + kc]);
    int vd = tid >> 1, vc = (tid & 1) * 16;    // V^T tile: 128 d x 32 keys
    unsigned vdst = lds_off(&Vs[vd * AT_LDV + vc]);

    v8f acc[8] = {};
    float mprev[8], lsum[8];
    for (int r = 0; r < 8; r++) { mprev[r] = -3.0e38f; lsum[r] = 0.f; }

    u16* Pw = Ps + wid * 16 * AT_LDP;
    int nj = qb * 4 + 4;                 // causal: key blocks of 32
    for (int j = 0; j < nj; j++) {
        {   // async load K tile
            const u16* src = Kc + (size_t)(j * 32 + kr) * Hh + (size_t)head * HDd + kc;
            async_b128(kdst,      src);
            async_b128(kdst + 16, src + 8);
        }
        {   // async load V^T tile
            const u16* src = Vt + ((size_t)head * HDd + vd) * Tt + j * 32 + vc;
            async_b128(vdst,      src);
            async_b128(vdst + 16, src + 8);
        }
        wait_async0();
        __syncthreads();

        // S = Q * K^T  (two 16x16 key tiles)
        v8f sts[2];
        for (int t = 0; t < 2; t++) {
            v8f s = {};
            for (int dc = 0; dc < 4; dc++) {
                Frag kf;
                int base = (t * 16 + mr) * AT_LDK + dc * 32 + half * 8;
                kf.q[0] = *(const uint4*)&Ks[base];
                kf.q[1] = *(const uint4*)&Ks[base + 16];
                s = wmma_bf16(qf[dc], kf, s);
            }
            sts[t] = s;
        }

        // online softmax update (rows distributed per C-layout)
        for (int r = 0; r < 8; r++) {
            int mg = qbase + r + half * 8;
            float v0 = sts[0][r] * scaling;
            float v1 = sts[1][r] * scaling;
            if (j * 32 + mr      > mg) v0 = -3.0e38f;
            if (j * 32 + 16 + mr > mg) v1 = -3.0e38f;
            float vmax = fmaxf(v0, v1);
            for (int off = 1; off < 16; off <<= 1)
                vmax = fmaxf(vmax, __shfl_xor(vmax, off));
            float mnew = fmaxf(mprev[r], vmax);
            float corr = __expf(mprev[r] - mnew);
            float p0 = __expf(v0 - mnew);
            float p1 = __expf(v1 - mnew);
            float rs = p0 + p1;
            for (int off = 1; off < 16; off <<= 1)
                rs += __shfl_xor(rs, off);
            lsum[r] = lsum[r] * corr + rs;
            mprev[r] = mnew;
            for (int t8 = 0; t8 < 8; t8++) acc[t8][r] *= corr;
            Pw[(r + half * 8) * AT_LDP + mr]      = f2bf(p0);
            Pw[(r + half * 8) * AT_LDP + 16 + mr] = f2bf(p1);
        }
        __syncthreads();   // P staging visible

        // O += P * V
        Frag pf;
        int pb = mr * AT_LDP + half * 8;
        pf.q[0] = *(const uint4*)&Pw[pb];
        pf.q[1] = *(const uint4*)&Pw[pb + 16];
        for (int t8 = 0; t8 < 8; t8++) {
            Frag vf;
            int vb = (t8 * 16 + mr) * AT_LDV + half * 8;
            vf.q[0] = *(const uint4*)&Vs[vb];
            vf.q[1] = *(const uint4*)&Vs[vb + 16];
            acc[t8] = wmma_bf16(pf, vf, acc[t8]);
        }
        __syncthreads();   // before next tile overwrite
    }

    for (int r = 0; r < 8; r++) {
        float inv = 1.0f / lsum[r];
        int m = qbase + r + half * 8;
        for (int t8 = 0; t8 < 8; t8++)
            Out[(size_t)m * Hh + (size_t)head * HDd + t8 * 16 + mr] =
                f2bf(acc[t8][r] * inv);
    }
}

// ---------------------------------------------------------------------------
// g = a1 * silu(a2)  (bf16 in, bf16 out)
// ---------------------------------------------------------------------------
__global__ __launch_bounds__(256) void silu_mul_kernel(
    const u16* __restrict__ A1, const u16* __restrict__ A2,
    u16* __restrict__ G)
{
    size_t i = (size_t)blockIdx.x * 256 + threadIdx.x;
    float a = bf2f(A1[i]);
    float b = bf2f(A2[i]);
    float s = b / (1.f + __expf(-b));
    G[i] = f2bf(a * s);
}

// ---------------------------------------------------------------------------
// Host-side orchestration
// ---------------------------------------------------------------------------
extern "C" void kernel_launch(void* const* d_in, const int* in_sizes, int n_in,
                              void* d_out, int out_size, void* d_ws, size_t ws_size,
                              hipStream_t stream)
{
    const int*   positions = (const int*)  d_in[0];
    const float* hidden    = (const float*)d_in[1];
    const float* ln1_scale = (const float*)d_in[2];
    const float* wq        = (const float*)d_in[3];
    const float* bq        = (const float*)d_in[4];
    const float* wk        = (const float*)d_in[5];
    const float* bk        = (const float*)d_in[6];
    const float* wv        = (const float*)d_in[7];
    const float* bv        = (const float*)d_in[8];
    const float* wo        = (const float*)d_in[9];
    const float* ln2_scale = (const float*)d_in[10];
    const float* w1        = (const float*)d_in[11];
    const float* w2        = (const float*)d_in[12];
    const float* wc        = (const float*)d_in[13];

    char* base = (char*)d_ws;
    size_t off = 0;
    auto alloc = [&](size_t bytes) -> char* {
        char* p = base + off;
        off += (bytes + 255) & ~(size_t)255;
        return p;
    };

    const size_t HH = (size_t)Hh * Hh;
    const size_t HF = (size_t)Hh * FFf;
    const size_t TH = (size_t)Tt * Hh;
    const size_t TF = (size_t)Tt * FFf;

    u16* wqT = (u16*)alloc(HH * 2);
    u16* wkT = (u16*)alloc(HH * 2);
    u16* wvT = (u16*)alloc(HH * 2);
    u16* woT = (u16*)alloc(HH * 2);
    u16* w1T = (u16*)alloc(HF * 2);
    u16* w2T = (u16*)alloc(HF * 2);
    u16* wcT = (u16*)alloc(HF * 2);
    u16* xn  = (u16*)alloc(TH * 2);
    u16* qb_ = (u16*)alloc(TH * 2);
    u16* kb_ = (u16*)alloc(TH * 2);
    u16* vb_ = (u16*)alloc(TH * 2);
    u16* vT_ = (u16*)alloc(TH * 2);
    u16* at_ = (u16*)alloc(TH * 2);
    float* h_ = (float*)alloc(TH * 4);
    u16* yn  = (u16*)alloc(TH * 2);
    u16* a1_ = (u16*)alloc(TF * 2);
    u16* a2_ = (u16*)alloc(TF * 2);
    u16* g_  = (u16*)alloc(TF * 2);
    (void)ws_size;

    dim3 tb32(32, 8);

    // --- weight convert + transpose (W[K][N] -> Wt[N][K], bf16) ---
    cvt_transpose_kernel<<<dim3(Hh / 32, Hh / 32), tb32, 0, stream>>>(wq, wqT, Hh, Hh);
    cvt_transpose_kernel<<<dim3(Hh / 32, Hh / 32), tb32, 0, stream>>>(wk, wkT, Hh, Hh);
    cvt_transpose_kernel<<<dim3(Hh / 32, Hh / 32), tb32, 0, stream>>>(wv, wvT, Hh, Hh);
    cvt_transpose_kernel<<<dim3(Hh / 32, Hh / 32), tb32, 0, stream>>>(wo, woT, Hh, Hh);
    cvt_transpose_kernel<<<dim3(FFf / 32, Hh / 32), tb32, 0, stream>>>(w1, w1T, Hh, FFf);
    cvt_transpose_kernel<<<dim3(FFf / 32, Hh / 32), tb32, 0, stream>>>(w2, w2T, Hh, FFf);
    cvt_transpose_kernel<<<dim3(Hh / 32, FFf / 32), tb32, 0, stream>>>(wc, wcT, FFf, Hh);

    // --- x = rmsnorm(hidden) ---
    rmsnorm_kernel<<<Tt, 256, 0, stream>>>(hidden, ln1_scale, xn, Hh);

    // --- q/k/v projections (bf16 out, +bias) ---
    dim3 gHH(Hh / 128, Tt / 128);
    gemm_bf16_kernel<<<gHH, 256, 0, stream>>>(xn, wqT, bq, nullptr, nullptr, qb_, Tt, Hh, Hh);
    gemm_bf16_kernel<<<gHH, 256, 0, stream>>>(xn, wkT, bk, nullptr, nullptr, kb_, Tt, Hh, Hh);
    gemm_bf16_kernel<<<gHH, 256, 0, stream>>>(xn, wvT, bv, nullptr, nullptr, vb_, Tt, Hh, Hh);

    // --- RoPE on q, k ---
    int rope_blocks = (Tt * NHh * (HDd / 2)) / 256;
    rope_kernel<<<rope_blocks, 256, 0, stream>>>(qb_, positions);
    rope_kernel<<<rope_blocks, 256, 0, stream>>>(kb_, positions);

    // --- V transpose per head ---
    transpose_v_kernel<<<dim3(Tt / 32, HDd / 32, NHh), tb32, 0, stream>>>(vb_, vT_);

    // --- flash attention ---
    attention_kernel<<<dim3(Tt / 128, NHh), 256, 0, stream>>>(qb_, kb_, vT_, at_);

    // --- h = hidden + attn @ wo (f32 out) ---
    gemm_bf16_kernel<<<gHH, 256, 0, stream>>>(at_, woT, nullptr, hidden, h_, nullptr, Tt, Hh, Hh);

    // --- y = rmsnorm(h) ---
    rmsnorm_kernel<<<Tt, 256, 0, stream>>>(h_, ln2_scale, yn, Hh);

    // --- a1 = y @ w1, a2 = y @ w2 (bf16 out) ---
    dim3 gHF(FFf / 128, Tt / 128);
    gemm_bf16_kernel<<<gHF, 256, 0, stream>>>(yn, w1T, nullptr, nullptr, nullptr, a1_, Tt, FFf, Hh);
    gemm_bf16_kernel<<<gHF, 256, 0, stream>>>(yn, w2T, nullptr, nullptr, nullptr, a2_, Tt, FFf, Hh);

    // --- g = a1 * silu(a2) ---
    silu_mul_kernel<<<(unsigned)(TF / 256), 256, 0, stream>>>(a1_, a2_, g_);

    // --- out = h + g @ wc (f32 out, residual) ---
    dim3 gFH(Hh / 128, Tt / 128);
    gemm_bf16_kernel<<<gFH, 256, 0, stream>>>(g_, wcT, nullptr, h_, (float*)d_out, nullptr, Tt, Hh, FFf);

    (void)in_sizes; (void)n_in; (void)out_size;
}
